// SimplifiedMultiHeadSelfAttention_62431644615170
// MI455X (gfx1250) — compile-verified, matrix operand order checked by reading, not run on hardware
//
#include <hip/hip_runtime.h>
#include <stdint.h>

// ---------------------------------------------------------------------------
// SimplifiedMultiHeadSelfAttention for MI455X (gfx1250, wave32, WMMA bf16)
//   z[4,2048,1024] -> QKV -> 64 x (2048x2048 attention, dh=64) -> out proj
// bf16 operands / f32 accumulate via v_wmma_f32_16x16x32_bf16.
// Tiles double-buffered in LDS via GLOBAL_LOAD_ASYNC_TO_LDS_B128 (ASYNCcnt),
// so global->LDS copies for step s+1 overlap WMMA compute on step s.
// ---------------------------------------------------------------------------

#define DMODEL 1024
#define NHEAD  16
#define DHEAD  64
#define SEQ    2048
#define BATCH  4
#define ROWS   (BATCH * SEQ)   // 8192
#define NQKV   (3 * DMODEL)    // 3072
#define GK     64              // GEMM K-step

typedef __bf16 bf16_t;
typedef __attribute__((ext_vector_type(16))) __bf16 bf16x16;
typedef __attribute__((ext_vector_type(8)))  __bf16 bf16x8;
typedef __attribute__((ext_vector_type(8)))  float  f32x8;

// Native converts on gfx1250 (v_cvt_*bf16*), one VALU op instead of 3.
static __device__ __forceinline__ bf16_t f32_to_bf16(float f) {
  return (bf16_t)f;
}

static __device__ __forceinline__ f32x8 zero8() {
  f32x8 z;
#pragma unroll
  for (int i = 0; i < 8; ++i) z[i] = 0.0f;
  return z;
}

// Async copy of one 16B chunk, global -> LDS (ASYNCcnt-tracked).
// VDST = LDS byte address (low 32 bits of generic shared pointer = LDS offset
// per the aperture mapping), VADDR = 64-bit global VA.
static __device__ __forceinline__ void copy16_g2lds(bf16_t* ldsdst,
                                                    const bf16_t* src) {
  uint32_t loff = (uint32_t)(uintptr_t)ldsdst;
  asm volatile("global_load_async_to_lds_b128 %0, %1, off"
               :: "v"(loff), "v"(src) : "memory");
}

static __device__ __forceinline__ void wait_async0() {
  asm volatile("s_wait_asynccnt 0x0" ::: "memory");
}

// A/B fragment of a 16x32 bf16 tile (row-major, K-contiguous source rows).
// ISA layout: lane half h: elems 0..7 <-> K = 8h+0..7, elems 8..15 <-> K=16+8h+0..7
static __device__ __forceinline__ bf16x16 load_frag(const bf16_t* row, int half) {
  bf16x8 lo = *(const bf16x8*)(row + half * 8);
  bf16x8 hi = *(const bf16x8*)(row + 16 + half * 8);
  bf16x16 f;
#pragma unroll
  for (int i = 0; i < 8; ++i) { f[i] = lo[i]; f[i + 8] = hi[i]; }
  return f;
}

static __device__ __forceinline__ f32x8 wmma_bf16(bf16x16 a, bf16x16 b, f32x8 c) {
  return __builtin_amdgcn_wmma_f32_16x16x32_bf16(false, a, false, b, (short)0, c,
                                                 false, false);
}

// ---------------------------------------------------------------------------
// Conversion kernels (one-shot; bandwidth trivial, results L2-resident)
// ---------------------------------------------------------------------------
__global__ void cvt_f32_bf16(const float* __restrict__ in, bf16_t* __restrict__ out,
                             int n) {
  int i = blockIdx.x * blockDim.x + threadIdx.x;
  if (i < n) out[i] = f32_to_bf16(in[i]);
}

// in: f32 [K][N] -> out: bf16 [N][K]  (all GEMMs become uniform A * B^T)
__global__ void transpose_f32_bf16(const float* __restrict__ in,
                                   bf16_t* __restrict__ out, int K, int N) {
  int i = blockIdx.x * blockDim.x + threadIdx.x;
  if (i < K * N) {
    int k = i / N, n = i - k * N;
    out[(size_t)n * K + k] = f32_to_bf16(in[i]);
  }
}

// ---------------------------------------------------------------------------
// Shared GEMM mainloop: block tile 128(M) x 256(N), K-step 64, double-buffered
// LDS (A: 2x16KB, B: 2x32KB). 8 waves as 2(M) x 4(N); wave tile 64x64 =
// 32 WMMA per K-step per wave.
// ---------------------------------------------------------------------------
static __device__ __forceinline__ void gemm_stage(
    const bf16_t* __restrict__ A, const bf16_t* __restrict__ B, int Kdim,
    bf16_t* ldsA, bf16_t* ldsB, int mblk, int nblk, int kb, int buf) {
  const int tid = threadIdx.x;
  bf16_t* dA = ldsA + buf * (128 * GK);
  bf16_t* dB = ldsB + buf * (256 * GK);
  // A tile 128x64 (16 KB = 1024 chunks, 4/thread)
#pragma unroll
  for (int i = 0; i < 4; ++i) {
    int c = tid + i * 256, row = c >> 3, part = c & 7;
    copy16_g2lds(dA + row * GK + part * 8,
                 A + (size_t)(mblk + row) * Kdim + kb + part * 8);
  }
  // B tile 256x64 (32 KB = 2048 chunks, 8/thread)
#pragma unroll
  for (int i = 0; i < 8; ++i) {
    int c = tid + i * 256, row = c >> 3, part = c & 7;
    copy16_g2lds(dB + row * GK + part * 8,
                 B + (size_t)(nblk + row) * Kdim + kb + part * 8);
  }
}

static __device__ __forceinline__ void gemm_mainloop(
    const bf16_t* __restrict__ A, const bf16_t* __restrict__ B, int Kdim,
    bf16_t* ldsA, bf16_t* ldsB, int mblk, int nblk, f32x8 acc[4][4]) {
  const int lane = threadIdx.x & 31;
  const int w    = threadIdx.x >> 5;
  const int half = lane >> 4;
  const int lm   = lane & 15;
  const int wm   = w & 1;        // 0..1  (M)
  const int wn   = w >> 1;       // 0..3  (N)

  gemm_stage(A, B, Kdim, ldsA, ldsB, mblk, nblk, 0, 0);
  const int nsteps = Kdim / GK;
  for (int s = 0; s < nsteps; ++s) {
    wait_async0();      // current buffer's copies landed
    __syncthreads();    // all waves done reading the buffer we stage next
    if (s + 1 < nsteps)
      gemm_stage(A, B, Kdim, ldsA, ldsB, mblk, nblk, (s + 1) * GK, (s + 1) & 1);
    const bf16_t* lA = ldsA + (s & 1) * (128 * GK);
    const bf16_t* lB = ldsB + (s & 1) * (256 * GK);
#pragma unroll
    for (int kc = 0; kc < 2; ++kc) {
      bf16x16 af[4], bfr[4];
#pragma unroll
      for (int mi = 0; mi < 4; ++mi)
        af[mi] = load_frag(lA + (wm * 64 + mi * 16 + lm) * GK + kc * 32, half);
#pragma unroll
      for (int ni = 0; ni < 4; ++ni)
        bfr[ni] = load_frag(lB + (wn * 64 + ni * 16 + lm) * GK + kc * 32, half);
#pragma unroll
      for (int ni = 0; ni < 4; ++ni)
#pragma unroll
        for (int mi = 0; mi < 4; ++mi)
          acc[mi][ni] = wmma_bf16(af[mi], bfr[ni], acc[mi][ni]);
    }
  }
}

// ---------------------------------------------------------------------------
// QKV projection: [8192,1024] x Wqkv^T[3072,1024] + b, scattered head-major:
// Q[bh][n][64] (pre-scaled by 1/sqrt(dh)), K[bh][n][64], Vt[bh][64][n].
// ---------------------------------------------------------------------------
__global__ __launch_bounds__(256) void qkv_gemm(
    const bf16_t* __restrict__ Zb, const bf16_t* __restrict__ Wt,
    const float* __restrict__ bias, bf16_t* __restrict__ Qb,
    bf16_t* __restrict__ Kb, bf16_t* __restrict__ Vt) {
  __shared__ bf16_t ldsA[2 * 128 * GK];
  __shared__ bf16_t ldsB[2 * 256 * GK];
  const int lane = threadIdx.x & 31;
  const int w    = threadIdx.x >> 5;
  const int half = lane >> 4;
  const int lm   = lane & 15;
  const int wm   = w & 1, wn = w >> 1;
  const int mblk = blockIdx.y * 128;
  const int nblk = blockIdx.x * 256;

  f32x8 acc[4][4];
#pragma unroll
  for (int i = 0; i < 4; ++i)
#pragma unroll
    for (int j = 0; j < 4; ++j) acc[i][j] = zero8();

  gemm_mainloop(Zb, Wt, DMODEL, ldsA, ldsB, mblk, nblk, acc);

#pragma unroll
  for (int ni = 0; ni < 4; ++ni) {
    const int c = nblk + wn * 64 + ni * 16 + lm;   // output column in [0,3072)
    const float bv = bias[c];
    const int h = c / 192;
    const int j = c - h * 192;
    const int part = j >> 6;                       // 0=Q 1=K 2=V
    const int d = j & 63;
    const float scl = (part == 0) ? 0.125f : 1.0f; // fold 1/sqrt(64) into Q
#pragma unroll
    for (int mi = 0; mi < 4; ++mi) {
#pragma unroll
      for (int r = 0; r < 8; ++r) {
        const int m    = mblk + wm * 64 + mi * 16 + half * 8 + r;
        const int bidx = m >> 11;                  // / SEQ
        const int nrow = m & (SEQ - 1);
        const bf16_t v = f32_to_bf16((acc[mi][ni][r] + bv) * scl);
        const int bh = bidx * NHEAD + h;
        if (part == 0)
          Qb[((size_t)bh * SEQ + nrow) * DHEAD + d] = v;
        else if (part == 1)
          Kb[((size_t)bh * SEQ + nrow) * DHEAD + d] = v;
        else
          Vt[((size_t)bh * DHEAD + d) * SEQ + nrow] = v;
      }
    }
  }
}

// ---------------------------------------------------------------------------
// Attention: block = 128 q rows of one (b,h); wave = 16 q rows.
// Per 32-key step: K tile (32x64) + V tile (64x32), double-buffered in LDS,
// shared by all 8 waves; 4 WMMA for S, online softmax (shfl_xor over 16-lane
// halves), P relaid out C->A through a per-wave LDS patch, 4 WMMA for P*V.
// ---------------------------------------------------------------------------
static __device__ __forceinline__ void attn_stage(
    const bf16_t* __restrict__ kbase, const bf16_t* __restrict__ vbase,
    bf16_t* ldsK, bf16_t* ldsV, int kb, int buf) {
  const int tid = threadIdx.x;
  // K tile: 32 rows x 64 d = 4 KB contiguous (256 chunks, 1/thread)
  copy16_g2lds(ldsK + buf * (32 * 64) + tid * 8,
               kbase + (size_t)kb * DHEAD + tid * 8);
  // V tile: 64 rows (stride SEQ) x 32 nk = 4 KB (1 chunk/thread)
  copy16_g2lds(ldsV + buf * (64 * 32) + (tid >> 2) * 32 + (tid & 3) * 8,
               vbase + (size_t)(tid >> 2) * SEQ + kb + (tid & 3) * 8);
}

__global__ __launch_bounds__(256) void attn_kernel(
    const bf16_t* __restrict__ Qb, const bf16_t* __restrict__ Kb,
    const bf16_t* __restrict__ Vt, bf16_t* __restrict__ An) {
  __shared__ bf16_t ldsK[2 * 32 * 64];   // rows k' (32), cols d (64)
  __shared__ bf16_t ldsV[2 * 64 * 32];   // rows d (64), cols nk (32)
  __shared__ bf16_t ldsP[8 * 16 * 32];   // per-wave P patch
  const int tid  = threadIdx.x;
  const int lane = tid & 31;
  const int w    = tid >> 5;
  const int half = lane >> 4;
  const int lm   = lane & 15;
  const int bh   = blockIdx.y;
  const int bidx = bh >> 4;
  const int h    = bh & 15;
  const int qbase = blockIdx.x * 128 + w * 16;

  const bf16_t* qrow  = Qb + ((size_t)bh * SEQ + qbase + lm) * DHEAD;
  const bf16_t* kbase = Kb + (size_t)bh * SEQ * DHEAD;
  const bf16_t* vbase = Vt + (size_t)bh * DHEAD * SEQ;

  bf16x16 qf[2];
  qf[0] = load_frag(qrow, half);
  qf[1] = load_frag(qrow + 32, half);

  f32x8 o[4];
#pragma unroll
  for (int t = 0; t < 4; ++t) o[t] = zero8();
  float mrow[8], lrow[8];
#pragma unroll
  for (int r = 0; r < 8; ++r) { mrow[r] = -1e30f; lrow[r] = 0.0f; }

  bf16_t* wl = ldsP + w * (16 * 32);

  attn_stage(kbase, vbase, ldsK, ldsV, 0, 0);
  const int nsteps = SEQ / 32;
  for (int s = 0; s < nsteps; ++s) {
    wait_async0();
    __syncthreads();
    if (s + 1 < nsteps)
      attn_stage(kbase, vbase, ldsK, ldsV, (s + 1) * 32, (s + 1) & 1);
    const bf16_t* lK = ldsK + (s & 1) * (32 * 64);
    const bf16_t* lV = ldsV + (s & 1) * (64 * 32);

    f32x8 sc[2];
#pragma unroll
    for (int t = 0; t < 2; ++t) {
      f32x8 st = zero8();
#pragma unroll
      for (int c = 0; c < 2; ++c) {
        bf16x16 kf = load_frag(lK + (t * 16 + lm) * 64 + c * 32, half);
        st = wmma_bf16(qf[c], kf, st);   // Q pre-scaled by 1/sqrt(dh)
      }
      sc[t] = st;
    }

    float alpha[8], p0[8], p1[8];
#pragma unroll
    for (int r = 0; r < 8; ++r) {
      float cm = fmaxf(sc[0][r], sc[1][r]);
      cm = fmaxf(cm, __shfl_xor(cm, 1, 32));
      cm = fmaxf(cm, __shfl_xor(cm, 2, 32));
      cm = fmaxf(cm, __shfl_xor(cm, 4, 32));
      cm = fmaxf(cm, __shfl_xor(cm, 8, 32));
      const float nm = fmaxf(mrow[r], cm);
      alpha[r] = __expf(mrow[r] - nm);
      mrow[r]  = nm;
      p0[r] = __expf(sc[0][r] - nm);
      p1[r] = __expf(sc[1][r] - nm);
      float rs = p0[r] + p1[r];
      rs += __shfl_xor(rs, 1, 32);
      rs += __shfl_xor(rs, 2, 32);
      rs += __shfl_xor(rs, 4, 32);
      rs += __shfl_xor(rs, 8, 32);
      lrow[r] = lrow[r] * alpha[r] + rs;
    }
#pragma unroll
    for (int t = 0; t < 4; ++t)
#pragma unroll
      for (int r = 0; r < 8; ++r) o[t][r] *= alpha[r];

    // P: C-layout -> A-layout via per-wave LDS patch (DS in-order per wave)
#pragma unroll
    for (int r = 0; r < 8; ++r) {
      wl[(half * 8 + r) * 32 + lm]      = f32_to_bf16(p0[r]);
      wl[(half * 8 + r) * 32 + 16 + lm] = f32_to_bf16(p1[r]);
    }
    bf16x16 pf = load_frag(wl + lm * 32, half);
#pragma unroll
    for (int t = 0; t < 4; ++t) {
      bf16x16 vf = load_frag(lV + (t * 16 + lm) * 32, half);
      o[t] = wmma_bf16(pf, vf, o[t]);
    }
  }

#pragma unroll
  for (int r = 0; r < 8; ++r) {
    const float inv = 1.0f / lrow[r];
    const int row = bidx * SEQ + qbase + half * 8 + r;
#pragma unroll
    for (int t = 0; t < 4; ++t)
      An[(size_t)row * DMODEL + h * DHEAD + t * 16 + lm] =
          f32_to_bf16(o[t][r] * inv);
  }
}

// ---------------------------------------------------------------------------
// Output projection: attn[8192,1024] x Wmsa^T[1024,1024] + b -> f32 d_out
// ---------------------------------------------------------------------------
__global__ __launch_bounds__(256) void msa_gemm(
    const bf16_t* __restrict__ Ab, const bf16_t* __restrict__ Wt,
    const float* __restrict__ bias, float* __restrict__ out) {
  __shared__ bf16_t ldsA[2 * 128 * GK];
  __shared__ bf16_t ldsB[2 * 256 * GK];
  const int lane = threadIdx.x & 31;
  const int w    = threadIdx.x >> 5;
  const int half = lane >> 4;
  const int lm   = lane & 15;
  const int wm   = w & 1, wn = w >> 1;
  const int mblk = blockIdx.y * 128;
  const int nblk = blockIdx.x * 256;

  f32x8 acc[4][4];
#pragma unroll
  for (int i = 0; i < 4; ++i)
#pragma unroll
    for (int j = 0; j < 4; ++j) acc[i][j] = zero8();

  gemm_mainloop(Ab, Wt, DMODEL, ldsA, ldsB, mblk, nblk, acc);

#pragma unroll
  for (int ni = 0; ni < 4; ++ni) {
    const int c = nblk + wn * 64 + ni * 16 + lm;
    const float bv = bias[c];
#pragma unroll
    for (int mi = 0; mi < 4; ++mi) {
#pragma unroll
      for (int r = 0; r < 8; ++r) {
        const int m = mblk + wm * 64 + mi * 16 + half * 8 + r;
        out[(size_t)m * DMODEL + c] = acc[mi][ni][r] + bv;
      }
    }
  }
}

// ---------------------------------------------------------------------------
extern "C" void kernel_launch(void* const* d_in, const int* in_sizes, int n_in,
                              void* d_out, int out_size, void* d_ws, size_t ws_size,
                              hipStream_t stream) {
  (void)in_sizes; (void)n_in; (void)out_size; (void)ws_size;
  const float* z    = (const float*)d_in[0];
  const float* Wqkv = (const float*)d_in[1];
  const float* bqkv = (const float*)d_in[2];
  const float* Wmsa = (const float*)d_in[3];
  const float* bmsa = (const float*)d_in[4];
  float* out = (float*)d_out;

  char* ws = (char*)d_ws;                               // 88 MiB used
  bf16_t* Zb    = (bf16_t*)(ws);                        // 16 MiB [8192][1024]
  bf16_t* Wqkvt = (bf16_t*)(ws + (16ull << 20));        //  6 MiB [3072][1024]
  bf16_t* Wmsat = (bf16_t*)(ws + (22ull << 20));        //  2 MiB [1024][1024]
  bf16_t* Qb    = (bf16_t*)(ws + (24ull << 20));        // 16 MiB [64][2048][64]
  bf16_t* Kb    = (bf16_t*)(ws + (40ull << 20));        // 16 MiB [64][2048][64]
  bf16_t* Vt    = (bf16_t*)(ws + (56ull << 20));        // 16 MiB [64][64][2048]
  bf16_t* An    = (bf16_t*)(ws + (72ull << 20));        // 16 MiB [8192][1024]

  cvt_f32_bf16<<<(ROWS * DMODEL) / 256, 256, 0, stream>>>(z, Zb, ROWS * DMODEL);
  transpose_f32_bf16<<<(DMODEL * NQKV) / 256, 256, 0, stream>>>(Wqkv, Wqkvt,
                                                                DMODEL, NQKV);
  transpose_f32_bf16<<<(DMODEL * DMODEL) / 256, 256, 0, stream>>>(Wmsa, Wmsat,
                                                                  DMODEL, DMODEL);
  qkv_gemm<<<dim3(NQKV / 256, ROWS / 128), 256, 0, stream>>>(Zb, Wqkvt, bqkv,
                                                             Qb, Kb, Vt);
  attn_kernel<<<dim3(SEQ / 128, BATCH * NHEAD), 256, 0, stream>>>(Qb, Kb, Vt, An);
  msa_gemm<<<dim3(DMODEL / 256, ROWS / 128), 256, 0, stream>>>(An, Wmsat, bmsa, out);
}
